// CustomCausalSelfAttention_5506148074179
// MI455X (gfx1250) — compile-verified, matrix operand order checked by reading, not run on hardware
//
#include <hip/hip_runtime.h>

// ---------------------------------------------------------------------------
// CDNA5 (gfx1250) bf16 WMMA implementation of MLA-style causal attention.
// wave32 only; WMMA D = A(16x32 bf16) * B(32x16 bf16) + C(16x16 f32).
// ---------------------------------------------------------------------------

typedef __attribute__((ext_vector_type(16))) __bf16 v16bf;
typedef __attribute__((ext_vector_type(8)))  float  v8f;

#define WMMA_BF16(A_, B_, C_) \
  __builtin_amdgcn_wmma_f32_16x16x32_bf16(false, (A_), false, (B_), (short)0, (C_), false, false)

union BF16Frag { v16bf v; unsigned u[8]; };

// k index for packed dword j (0..7) of an A/B fragment, per the ISA 16-bit
// 16x32 layout: lanes 0-15 hold K 0..7 (vgpr0-3) & 16..23 (vgpr4-7);
// lanes 16-31 hold K 8..15 & 24..31. (hi = lane>>4)
__device__ __forceinline__ int frag_k(int j, int hi) {
  return ((j & 4) << 2) + (hi << 3) + ((j & 3) << 1);
}

#define D_MODEL 2048
#define NH      32
#define DQK     64

// ---------------------------------------------------------------------------
// Generic bf16 GEMM: C[M,N] = A[M,K] @ B[K,N], fp32 accumulate.
// Block tile 128x64, 8 waves (4x2), wave tile 32x32, BK=32 (one WMMA k-step).
// ---------------------------------------------------------------------------
namespace gemmcfg {
constexpr int BM = 128, BN = 64, BK = 32;
constexpr int LSTR = BK + 8; // padded LDS row stride (elements)
}

template<bool BF16OUT>
__global__ void __launch_bounds__(256)
gemm_bf16(const __bf16* __restrict__ A, const __bf16* __restrict__ Bw,
          __bf16* __restrict__ Cb, float* __restrict__ Cf,
          int M, int N, int K)
{
  using namespace gemmcfg;
  __shared__ __bf16 sA[BM * LSTR];
  __shared__ __bf16 sB[BN * LSTR];   // transposed: [n][k]

  const int tid  = threadIdx.x;
  const int lane = tid & 31, wave = tid >> 5;
  const int wm = wave >> 1, wn = wave & 1;
  const int bm = blockIdx.y * BM, bn = blockIdx.x * BN;
  const int r  = lane & 15, hi = lane >> 4;

  v8f acc[2][2];
  #pragma unroll
  for (int mt = 0; mt < 2; ++mt)
    #pragma unroll
    for (int nt = 0; nt < 2; ++nt)
      #pragma unroll
      for (int v = 0; v < 8; ++v) acc[mt][nt][v] = 0.f;

  const int arow = tid >> 1, ak = (tid & 1) * 16;   // A: 16 bf16 per thread
  const int bcol = tid & 63, bk = (tid >> 6) * 8;   // B: 8 bf16 per thread (transposed)

  for (int k0 = 0; k0 < K; k0 += BK) {
    // ---- stage A tile (row-major) ----
    const uint4* ag = (const uint4*)(A + (size_t)(bm + arow) * K + k0 + ak);
    uint4 a0 = ag[0], a1 = ag[1];
    *(uint4*)&sA[arow * LSTR + ak]     = a0;
    *(uint4*)&sA[arow * LSTR + ak + 8] = a1;
    // ---- stage B tile transposed to [n][k] ----
    union { unsigned short s[8]; uint4 v; } bt;
    #pragma unroll
    for (int j = 0; j < 8; ++j)
      bt.s[j] = *(const unsigned short*)(Bw + (size_t)(k0 + bk + j) * N + bn + bcol);
    *(uint4*)&sB[bcol * LSTR + bk] = bt.v;
    __syncthreads();

    // ---- fragments + WMMA ----
    BF16Frag af[2], bfg[2];
    #pragma unroll
    for (int mt = 0; mt < 2; ++mt) {
      const char* base = (const char*)&sA[(wm * 32 + mt * 16 + r) * LSTR];
      #pragma unroll
      for (int j = 0; j < 8; ++j)
        af[mt].u[j] = *(const unsigned*)(base + frag_k(j, hi) * 2);
    }
    #pragma unroll
    for (int nt = 0; nt < 2; ++nt) {
      const char* base = (const char*)&sB[(wn * 32 + nt * 16 + r) * LSTR];
      #pragma unroll
      for (int j = 0; j < 8; ++j)
        bfg[nt].u[j] = *(const unsigned*)(base + frag_k(j, hi) * 2);
    }
    #pragma unroll
    for (int mt = 0; mt < 2; ++mt)
      #pragma unroll
      for (int nt = 0; nt < 2; ++nt)
        acc[mt][nt] = WMMA_BF16(af[mt].v, bfg[nt].v, acc[mt][nt]);
    __syncthreads();
  }

  // ---- epilogue (C layout: vgpr v -> row v + 8*hi, col = lane&15) ----
  #pragma unroll
  for (int mt = 0; mt < 2; ++mt)
    #pragma unroll
    for (int nt = 0; nt < 2; ++nt)
      #pragma unroll
      for (int v = 0; v < 8; ++v) {
        const int row = bm + wm * 32 + mt * 16 + hi * 8 + v;
        const int col = bn + wn * 32 + nt * 16 + r;
        if (BF16OUT) Cb[(size_t)row * N + col] = (__bf16)acc[mt][nt][v];
        else         Cf[(size_t)row * N + col] = acc[mt][nt][v];
      }
}

// ---------------------------------------------------------------------------
// Flash attention: one block per (q-tile of 128, head, batch). 8 waves x 16 rows.
// Q rows (s*B+b) layout; output rows (b*S+s) layout for the final GEMM.
// ---------------------------------------------------------------------------
constexpr int FQ = 128;
constexpr int FK = 64;
constexpr int KSTR = 72;

__global__ void __launch_bounds__(256)
flash_attn(const __bf16* __restrict__ Q, const __bf16* __restrict__ Km,
           const __bf16* __restrict__ Vm, const unsigned char* __restrict__ mask,
           __bf16* __restrict__ O, int S, int B)
{
  __shared__ __bf16 sK[FK * KSTR];    // [key][d]
  __shared__ __bf16 sVt[FK * KSTR];   // [d][key]
  __shared__ float  sS[FQ * FK];      // fp32 scores; bf16 P overlays row starts
  __shared__ float  sM[FQ], sL[FQ], sC[FQ];

  const int tid = threadIdx.x, lane = tid & 31, wave = tid >> 5;
  const int r = lane & 15, hi = lane >> 4;
  const int q0 = blockIdx.x * FQ;
  const int h  = blockIdx.y;
  const int b  = blockIdx.z;
  const float scale = 0.125f; // 1/sqrt(64)

  if (tid < FQ) { sM[tid] = -1e30f; sL[tid] = 0.f; }

  // Q fragments (A-matrix) straight from global, held for the whole block.
  BF16Frag aq[2];
  {
    const int qs = q0 + wave * 16 + r;
    const char* qrow = (const char*)(Q + (size_t)(qs * B + b) * D_MODEL + h * DQK);
    #pragma unroll
    for (int f = 0; f < 2; ++f)
      #pragma unroll
      for (int j = 0; j < 8; ++j)
        aq[f].u[j] = *(const unsigned*)(qrow + (f * 32 + frag_k(j, hi)) * 2);
  }

  v8f accO[4];
  #pragma unroll
  for (int nt = 0; nt < 4; ++nt)
    #pragma unroll
    for (int v = 0; v < 8; ++v) accO[nt][v] = 0.f;

  const int qmax = q0 + FQ - 1;
  const int ntiles = qmax / FK + 1;   // causal: only tiles up to the diagonal
  __syncthreads();

  for (int t = 0; t < ntiles; ++t) {
    const int key0 = t * FK;
    { // stage K tile [key][d]
      const int krow = tid >> 2, ko = (tid & 3) * 16;
      const uint4* g = (const uint4*)(Km + (size_t)((key0 + krow) * B + b) * D_MODEL + h * DQK + ko);
      uint4 x0 = g[0], x1 = g[1];
      *(uint4*)&sK[krow * KSTR + ko]     = x0;
      *(uint4*)&sK[krow * KSTR + ko + 8] = x1;
    }
    { // stage V tile transposed [d][key]
      const int d = tid & 63, kk0 = (tid >> 6) * 16;
      union { unsigned short s[16]; uint4 v[2]; } vt;
      #pragma unroll
      for (int j = 0; j < 16; ++j)
        vt.s[j] = *(const unsigned short*)(Vm + (size_t)((key0 + kk0 + j) * B + b) * D_MODEL + h * DQK + d);
      *(uint4*)&sVt[d * KSTR + kk0]     = vt.v[0];
      *(uint4*)&sVt[d * KSTR + kk0 + 8] = vt.v[1];
    }
    __syncthreads();

    // ---- S = scale * Q K^T for this wave's 16 rows x 64 keys ----
    #pragma unroll
    for (int nt = 0; nt < 4; ++nt) {
      BF16Frag bk[2];
      const char* kb = (const char*)&sK[(nt * 16 + r) * KSTR];
      #pragma unroll
      for (int f = 0; f < 2; ++f)
        #pragma unroll
        for (int j = 0; j < 8; ++j)
          bk[f].u[j] = *(const unsigned*)(kb + (f * 32 + frag_k(j, hi)) * 2);
      v8f sc;
      #pragma unroll
      for (int v = 0; v < 8; ++v) sc[v] = 0.f;
      sc = WMMA_BF16(aq[0].v, bk[0].v, sc);
      sc = WMMA_BF16(aq[1].v, bk[1].v, sc);
      #pragma unroll
      for (int v = 0; v < 8; ++v)
        sS[(wave * 16 + hi * 8 + v) * FK + nt * 16 + r] = sc[v] * scale;
    }
    __syncthreads();

    // ---- per-row online softmax update (threads 0..127, one row each) ----
    if (tid < FQ) {
      const int row = tid, qs = q0 + row;
      float* srow = &sS[row * FK];
      const float mOld = sM[row];
      float tmax = -1e30f;
      for (int i = 0; i < FK; ++i) {
        const int ks = key0 + i;
        const float sv = ((ks <= qs) && mask[b * S + ks]) ? srow[i] : -1e9f;
        tmax = fmaxf(tmax, sv);
      }
      const float mNew = fmaxf(mOld, tmax);
      const float corr = __expf(mOld - mNew);
      float lsum = 0.f;
      __bf16* prow = (__bf16*)srow;   // bf16 P overlays first half of fp32 row
      for (int i = 0; i < FK; ++i) {
        const int ks = key0 + i;
        const float sv = ((ks <= qs) && mask[b * S + ks]) ? srow[i] : -1e9f;
        const float p = __expf(sv - mNew);
        lsum += p;
        prow[i] = (__bf16)p;          // safe: fp32[i/2] already consumed
      }
      sM[row] = mNew;
      sL[row] = sL[row] * corr + lsum;
      sC[row] = corr;
    }
    __syncthreads();

    // ---- O = O*corr + P @ V ----
    {
      float cv[8];
      #pragma unroll
      for (int v = 0; v < 8; ++v) cv[v] = sC[wave * 16 + hi * 8 + v];
      #pragma unroll
      for (int nt = 0; nt < 4; ++nt)
        #pragma unroll
        for (int v = 0; v < 8; ++v) accO[nt][v] *= cv[v];

      BF16Frag ap[2];
      const char* pb = (const char*)&sS[(wave * 16 + r) * FK];
      #pragma unroll
      for (int f = 0; f < 2; ++f)
        #pragma unroll
        for (int j = 0; j < 8; ++j)
          ap[f].u[j] = *(const unsigned*)(pb + (f * 32 + frag_k(j, hi)) * 2);

      #pragma unroll
      for (int nt = 0; nt < 4; ++nt) {
        BF16Frag bv[2];
        const char* vb = (const char*)&sVt[(nt * 16 + r) * KSTR];
        #pragma unroll
        for (int f = 0; f < 2; ++f)
          #pragma unroll
          for (int j = 0; j < 8; ++j)
            bv[f].u[j] = *(const unsigned*)(vb + (f * 32 + frag_k(j, hi)) * 2);
        accO[nt] = WMMA_BF16(ap[0].v, bv[0].v, accO[nt]);
        accO[nt] = WMMA_BF16(ap[1].v, bv[1].v, accO[nt]);
      }
    }
    __syncthreads();
  }

  // ---- finalize: O /= l, apply sequence mask, store rows (b*S+s) ----
  #pragma unroll
  for (int v = 0; v < 8; ++v) {
    const int row = wave * 16 + hi * 8 + v;
    const int qs = q0 + row;
    const float l = sL[row];
    const float so = (l > 0.f ? 1.f / l : 0.f) * (mask[b * S + qs] ? 1.f : 0.f);
    #pragma unroll
    for (int nt = 0; nt < 4; ++nt)
      O[(size_t)(b * S + qs) * D_MODEL + h * DQK + nt * 16 + r] = (__bf16)(accO[nt][v] * so);
  }
}

// ---------------------------------------------------------------------------
// Small helper kernels
// ---------------------------------------------------------------------------
__global__ void cvt_f32_bf16(const float* __restrict__ in, __bf16* __restrict__ out, int n) {
  int i = blockIdx.x * blockDim.x + threadIdx.x;
  if (i < n) out[i] = (__bf16)in[i];
}

__global__ void pos_from_mask(const unsigned char* __restrict__ mask, int* __restrict__ pos,
                              int S, int B) {
  int idx = blockIdx.x * blockDim.x + threadIdx.x;
  if (idx >= B * S) return;
  const int b = idx / S, s = idx % S;
  int c = 0;
  for (int i = 0; i <= s; ++i) c += mask[b * S + i] ? 1 : 0;
  pos[idx] = c - 1;
}

// RoPE over full 64-dim head (half = 32), in place on bf16 Q (rows s*B+b).
__global__ void rope_q(__bf16* __restrict__ q, const int* __restrict__ pos, int S, int B) {
  const int total = S * B * NH * 32;
  int idx = blockIdx.x * blockDim.x + threadIdx.x;
  if (idx >= total) return;
  const int j = idx & 31, h = (idx >> 5) & 31, m = idx >> 10;
  const int b = m % B, s = m / B;
  const float p = (float)pos[b * S + s];
  const float freq = __expf(-((float)j * (1.f / 32.f)) * 9.210340371976184f); // ln(1e4)
  float sn, cs;
  __sincosf(p * freq, &sn, &cs);
  __bf16* base = q + (size_t)m * D_MODEL + h * DQK;
  const float x1 = (float)base[j], x2 = (float)base[j + 32];
  base[j]      = (__bf16)(x1 * cs - x2 * sn);
  base[j + 32] = (__bf16)(x2 * cs + x1 * sn);
}

// Assemble K from rope-part (Wkr, dims 0..15 of each head) and nope-part
// (Wuk, dims 16..63), then apply full-head RoPE. Rows s*B+b.
__global__ void build_rope_k(const __bf16* __restrict__ kr, const __bf16* __restrict__ kc,
                             __bf16* __restrict__ k, const int* __restrict__ pos, int S, int B) {
  const int total = S * B * NH * 32;
  int idx = blockIdx.x * blockDim.x + threadIdx.x;
  if (idx >= total) return;
  const int j = idx & 31, h = (idx >> 5) & 31, m = idx >> 10;
  const int b = m % B, s = m / B;
  const float p = (float)pos[b * S + s];
  const float freq = __expf(-((float)j * (1.f / 32.f)) * 9.210340371976184f);
  float sn, cs;
  __sincosf(p * freq, &sn, &cs);
  const float x1 = (j < 16) ? (float)kr[(size_t)m * 512 + h * 16 + j]
                            : (float)kc[(size_t)m * 1536 + h * 48 + (j - 16)];
  const float x2 = (float)kc[(size_t)m * 1536 + h * 48 + (j + 16)]; // (j+32)-16
  __bf16* base = k + (size_t)m * D_MODEL + h * DQK;
  base[j]      = (__bf16)(x1 * cs - x2 * sn);
  base[j + 32] = (__bf16)(x2 * cs + x1 * sn);
}

// ---------------------------------------------------------------------------
// Host orchestration
// ---------------------------------------------------------------------------
extern "C" void kernel_launch(void* const* d_in, const int* in_sizes, int n_in,
                              void* d_out, int out_size, void* d_ws, size_t ws_size,
                              hipStream_t stream)
{
  const float*         hidden = (const float*)d_in[0];
  const unsigned char* mask   = (const unsigned char*)d_in[1];
  const float* Wq  = (const float*)d_in[2];
  const float* Wkr = (const float*)d_in[3];
  const float* Wdk = (const float*)d_in[4];
  const float* Wuk = (const float*)d_in[5];
  const float* Wuv = (const float*)d_in[6];
  const float* Wo  = (const float*)d_in[7];
  float* out = (float*)d_out;

  const int SB = in_sizes[0] / D_MODEL;  // S*B = 2048
  const int B = 2, S = SB / B;
  const int LR = 512, KRN = 512, KCN = 1536, QN = NH * DQK; // 2048

  size_t off = 0;
  auto alloc = [&](size_t bytes) -> void* {
    off = (off + 255) & ~(size_t)255;
    void* p = (char*)d_ws + off;
    off += bytes;
    return p;
  };
  __bf16* hidB = (__bf16*)alloc((size_t)SB * D_MODEL * 2);
  __bf16* wqB  = (__bf16*)alloc((size_t)D_MODEL * QN * 2);
  __bf16* wkrB = (__bf16*)alloc((size_t)D_MODEL * KRN * 2);
  __bf16* wdkB = (__bf16*)alloc((size_t)D_MODEL * LR * 2);
  __bf16* wukB = (__bf16*)alloc((size_t)LR * KCN * 2);
  __bf16* wuvB = (__bf16*)alloc((size_t)LR * QN * 2);
  __bf16* woB  = (__bf16*)alloc((size_t)QN * D_MODEL * 2);
  __bf16* qB   = (__bf16*)alloc((size_t)SB * QN * 2);
  __bf16* krB  = (__bf16*)alloc((size_t)SB * KRN * 2);
  __bf16* ckvB = (__bf16*)alloc((size_t)SB * LR * 2);
  __bf16* kcB  = (__bf16*)alloc((size_t)SB * KCN * 2);
  __bf16* vB   = (__bf16*)alloc((size_t)SB * QN * 2);
  __bf16* kB   = (__bf16*)alloc((size_t)SB * QN * 2);
  __bf16* attB = (__bf16*)alloc((size_t)SB * QN * 2);
  int*    posA = (int*)alloc((size_t)SB * sizeof(int));

  auto cvt = [&](const float* src, __bf16* dst, int n) {
    cvt_f32_bf16<<<(n + 255) / 256, 256, 0, stream>>>(src, dst, n);
  };
  cvt(hidden, hidB, SB * D_MODEL);
  cvt(Wq,  wqB,  D_MODEL * QN);
  cvt(Wkr, wkrB, D_MODEL * KRN);
  cvt(Wdk, wdkB, D_MODEL * LR);
  cvt(Wuk, wukB, LR * KCN);
  cvt(Wuv, wuvB, LR * QN);
  cvt(Wo,  woB,  QN * D_MODEL);

  pos_from_mask<<<(SB + 255) / 256, 256, 0, stream>>>(mask, posA, S, B);

  auto gemmB = [&](const __bf16* A, const __bf16* Bw, __bf16* C, int M, int N, int K) {
    gemm_bf16<true><<<dim3(N / 64, M / 128), 256, 0, stream>>>(A, Bw, C, nullptr, M, N, K);
  };
  gemmB(hidB, wqB,  qB,   SB, QN,  D_MODEL);  // q
  gemmB(hidB, wkrB, krB,  SB, KRN, D_MODEL);  // k_r
  gemmB(hidB, wdkB, ckvB, SB, LR,  D_MODEL);  // c_kv
  gemmB(ckvB, wukB, kcB,  SB, KCN, LR);       // k_c
  gemmB(ckvB, wuvB, vB,   SB, QN,  LR);       // v

  const int ropeTotal = SB * NH * 32;
  rope_q<<<(ropeTotal + 255) / 256, 256, 0, stream>>>(qB, posA, S, B);
  build_rope_k<<<(ropeTotal + 255) / 256, 256, 0, stream>>>(krB, kcB, kB, posA, S, B);

  flash_attn<<<dim3(S / FQ, NH, B), 256, 0, stream>>>(qB, kB, vB, mask, attB, S, B);

  // out(B*S, D_MODEL) = attn(B*S, 2048) @ Wo(2048, 2048), fp32 store to d_out
  gemm_bf16<false><<<dim3(D_MODEL / 64, SB / 128), 256, 0, stream>>>(
      attB, woB, nullptr, out, SB, D_MODEL, D_MODEL);
}
